// RPN_85907935855331
// MI455X (gfx1250) — matrix operand order repaired and checked by reading, not compiled, not run on hardware
//
#include <hip/hip_runtime.h>
#include <math.h>

// ---------------- problem constants ----------------
static constexpr int NB   = 4;        // batch
static constexpr int CIN  = 256;
static constexpr int H    = 100;
static constexpr int W    = 152;
static constexpr int HW   = H * W;            // 15200
static constexpr int M    = NB * HW;          // 60800 spatial rows
static constexpr int AAN  = 15;               // anchors per position
static constexpr int ANCH = HW * AAN;         // 228000 per image
static constexpr int PRE  = 2000;
static constexpr int POST = 1000;
static constexpr float NMS_TH = 0.7f;
static constexpr float XFORM_CLIP = 4.135166556742356f; // log(1000/16)
static constexpr int KSTEPS_CONV = 72;        // (9*256)/32
static constexpr int COTILES = 16;            // 256/16
static constexpr int MTILES  = M / 16;        // 3800
static constexpr int CAND_MAX = 4096;
static constexpr int WHEADS_BYTES = 40960;    // (8 + 32) frags * 512 bf16 * 2B

// ---------------- vector types ----------------
typedef __attribute__((ext_vector_type(16))) __bf16 v16bf;
typedef __attribute__((ext_vector_type(8)))  float  v8f;
typedef __attribute__((ext_vector_type(4)))  unsigned int u32x4;
typedef __attribute__((ext_vector_type(8)))  int i32x8;
typedef __attribute__((ext_vector_type(4)))  int i32x4;

union AFrag { v16bf v; u32x4 q[2]; };

__device__ inline unsigned short f2bfbits(float f) {
    unsigned u = __float_as_uint(f);
    unsigned r = 0x7FFFu + ((u >> 16) & 1u);
    return (unsigned short)((u + r) >> 16);
}

__device__ inline v8f wmma_bf16(v16bf a, v16bf b, v8f c) {
    return __builtin_amdgcn_wmma_f32_16x16x32_bf16(false, a, false, b, (short)0, c, false, false);
}

// ---------------- utility ----------------
__global__ void k_zero32(unsigned* p, int n) {
    int i = blockIdx.x * blockDim.x + threadIdx.x;
    if (i < n) p[i] = 0u;
}

// fp32 NCHW -> bf16 NHWC
__global__ void k_cvt_feat(const float* __restrict__ f, unsigned short* __restrict__ featB) {
    int idx = blockIdx.x * blockDim.x + threadIdx.x;
    if (idx >= NB * HW * 256) return;
    int c = idx & 255;
    int pos = idx >> 8;
    int x = pos % W;
    int t = pos / W;
    int y = t % H;
    int n = t / H;
    featB[idx] = f2bfbits(f[(((size_t)n * 256 + c) * H + y) * W + x]);
}

// pack OIHW conv weights into WMMA B fragments: [t=0..71][coTile=0..15][lane=0..31][e=0..15]
// K-order: K = (r*3+s)*256 + ci ; B layout: n = lane&15, K = t*32 + ((lane>=16)?16:0) + e
__global__ void k_pack_conv(const float* __restrict__ w, unsigned short* __restrict__ wp) {
    int tid = blockIdx.x * blockDim.x + threadIdx.x;
    if (tid >= KSTEPS_CONV * COTILES * 32 * 16) return;
    int e    = tid & 15;
    int lane = (tid >> 4) & 31;
    int frag = tid >> 9;
    int coTile = frag & 15;
    int t      = frag >> 4;
    int nCol = coTile * 16 + (lane & 15);
    int K = t * 32 + ((lane >> 4) << 4) + e;
    int rs = K >> 8;
    int ci = K & 255;
    wp[tid] = f2bfbits(w[((size_t)nCol * 256 + ci) * 9 + rs]);
}

// pack cls (15x256 -> 16 cols, frags 0..7) and bbox (60x256 -> 64 cols, frags 8..39)
// into ONE contiguous buffer wheads (cls first, then bbox) for a single TDM staging DMA.
__global__ void k_pack_heads(const float* __restrict__ cls_w, const float* __restrict__ bbox_w,
                             unsigned short* __restrict__ wheads) {
    int tid = blockIdx.x * blockDim.x + threadIdx.x;
    if (tid >= (8 + 32) * 512) return;
    int e    = tid & 15;
    int lane = (tid >> 4) & 31;
    int frag = tid >> 9;
    int kl = ((lane >> 4) << 4) + e;  // local K in [0,32)
    float v;
    if (frag < 8) {
        int t = frag;
        int co = lane & 15;
        int ci = t * 32 + kl;
        v = (co < 15) ? cls_w[(size_t)co * 256 + ci] : 0.f;
    } else {
        int f  = frag - 8;          // f = t*4 + bt
        int bt = f & 3;
        int t  = f >> 2;
        int co = bt * 16 + (lane & 15);
        int ci = t * 32 + kl;
        v = (co < 60) ? bbox_w[(size_t)co * 256 + ci] : 0.f;
    }
    wheads[tid] = f2bfbits(v);
}

// 3x3 conv, SAME pad, implicit GEMM via WMMA.
// One wave = one 16(M)x64(Co) tile: 4 independent accumulator chains share each A fragment.
__global__ __launch_bounds__(256) void k_conv3x3(const unsigned short* __restrict__ featB,
                                                 const unsigned short* __restrict__ wpack,
                                                 const float* __restrict__ conv_b,
                                                 unsigned short* __restrict__ tB) {
    int wave = blockIdx.x * (blockDim.x >> 5) + (threadIdx.x >> 5);
    int lane = threadIdx.x & 31;
    if (wave >= MTILES * 4) return;
    int mTile = wave >> 2;
    int coQ   = wave & 3;                    // which group of 4 co-tiles
    int m = mTile * 16 + (lane & 15);
    int nImg = m / HW;
    int rem  = m - nImg * HW;
    int y = rem / W;
    int x = rem - y * W;
    int off = (lane >> 4) << 4;              // 0 or 16 (A-fragment K offset)
    const u32x4 ZQ = {0u, 0u, 0u, 0u};
    v8f acc[4];
#pragma unroll
    for (int q = 0; q < 4; ++q) acc[q] = (v8f){0.f, 0.f, 0.f, 0.f, 0.f, 0.f, 0.f, 0.f};

    for (int rs = 0; rs < 9; ++rs) {
        int r = rs / 3;
        int s = rs - 3 * r;
        int yy = y + r - 1, xx = x + s - 1;
        bool inb = ((unsigned)yy < (unsigned)H) && ((unsigned)xx < (unsigned)W);
        const unsigned short* abase =
            featB + (((size_t)((nImg * H + yy) * W + xx)) << 8) + off;
        // frag index = (rs*8 + c8)*16 + coQ*4 + q ; elem offset = frag*512 + lane*16
        const unsigned short* bbase =
            wpack + ((size_t)((rs * 8 * 16 + coQ * 4) * 32 + lane)) * 16;
#pragma unroll
        for (int c8 = 0; c8 < 8; ++c8) {
            AFrag a;
            if (inb) {
                const unsigned short* p = abase + (c8 << 5);   // +32 channels per step
                a.q[0] = *(const u32x4*)p;
                a.q[1] = *(const u32x4*)(p + 8);
            } else {
                a.q[0] = ZQ; a.q[1] = ZQ;
            }
#pragma unroll
            for (int q = 0; q < 4; ++q) {
                v16bf b = *(const v16bf*)(bbase + (size_t)c8 * 8192 + q * 512);
                acc[q] = wmma_bf16(a.v, b, acc[q]);
            }
        }
    }
#pragma unroll
    for (int q = 0; q < 4; ++q) {
        int co = (coQ * 4 + q) * 16 + (lane & 15);
        float bias = conv_b[co];
#pragma unroll
        for (int v = 0; v < 8; ++v) {
            int mr = mTile * 16 + v + ((lane >> 4) << 3);  // C/D layout: M = v (+8 upper half)
            float val = acc[q][v] + bias;
            val = val > 0.f ? val : 0.f;                   // ReLU
            tB[((size_t)mr << 8) + co] = f2bfbits(val);
        }
    }
}

// 1x1 heads: one wave per M-tile, 5 N-tiles (1 cls + 4 bbox), K = 256.
// Head weights staged into LDS once per block via the Tensor Data Mover.
__global__ __launch_bounds__(256) void k_heads(const unsigned short* __restrict__ tB,
                                               const unsigned short* __restrict__ wheads,
                                               const float* __restrict__ cls_b,
                                               const float* __restrict__ bbox_b,
                                               float* __restrict__ scores,
                                               float* __restrict__ breg) {
    __shared__ __align__(32) unsigned short sw[WHEADS_BYTES / 2];   // 40 KB of LDS

    // ---- TDM: one DMA per block copies all head-weight fragments into LDS ----
    if ((threadIdx.x >> 5) == 0) {
        unsigned lds_off = (unsigned)(unsigned long long)(void*)sw;
        unsigned long long ga = (unsigned long long)wheads;
        const unsigned NE = WHEADS_BYTES / 8;   // 5120 x 8-byte elements
        u32x4 g0;
        g0[0] = 1u;                                          // count=1, user descriptor
        g0[1] = lds_off;                                     // lds_addr (bytes)
        g0[2] = (unsigned)(ga & 0xFFFFFFFFu);                // global_addr[31:0]
        g0[3] = (unsigned)((ga >> 32) & 0x1FFFFFFu)          // global_addr[56:32]
              | (2u << 30);                                  // type=2 ("image")
        i32x8 g1;
        g1[0] = (int)(3u << 16);        // workgroup_mask=0, data_size=3 (8B)
        g1[1] = (int)(NE << 16);        // tensor_dim0[15:0] in bits[31:16]
        g1[2] = (int)(1u << 16);        // tensor_dim0[31:16]=0, tensor_dim1=1 (lo)
        g1[3] = (int)(NE << 16);        // tensor_dim1 hi=0, tile_dim0=NE
        g1[4] = 1;                      // tile_dim1=1, tile_dim2=0
        g1[5] = (int)NE;                // tensor_dim0_stride[31:0]
        g1[6] = 0;                      // stride hi / dim1_stride lo
        g1[7] = 0;
        i32x4 gz4 = {0, 0, 0, 0};
        i32x8 gz8 = {0, 0, 0, 0, 0, 0, 0, 0};
        __builtin_amdgcn_tensor_load_to_lds(g0, g1, gz4, gz4, gz8, 0);
        __builtin_amdgcn_s_wait_tensorcnt(0);
    }
    __syncthreads();

    int wave = blockIdx.x * (blockDim.x >> 5) + (threadIdx.x >> 5);
    int lane = threadIdx.x & 31;
    if (wave >= MTILES) return;
    int mTile = wave;
    int m = mTile * 16 + (lane & 15);
    int off = (lane >> 4) << 4;
    v8f acc[5];
#pragma unroll
    for (int i = 0; i < 5; ++i) acc[i] = (v8f){0.f, 0.f, 0.f, 0.f, 0.f, 0.f, 0.f, 0.f};
#pragma unroll
    for (int t = 0; t < 8; ++t) {
        int ci0 = t << 5;
        const unsigned short* base = tB + (((size_t)m) << 8) + ci0 + off;
        AFrag a;
        a.q[0] = *(const u32x4*)base;
        a.q[1] = *(const u32x4*)(base + 8);
        // cls frags at sw[0..8*512), bbox frags at sw[8*512 + ...)
        v16bf bc = *(const v16bf*)(&sw[((size_t)t * 32 + lane) * 16]);
        acc[0] = wmma_bf16(a.v, bc, acc[0]);
#pragma unroll
        for (int bt = 0; bt < 4; ++bt) {
            v16bf bb = *(const v16bf*)(&sw[(8 + (t << 2) + bt) * 512 + lane * 16]);
            acc[1 + bt] = wmma_bf16(a.v, bb, acc[1 + bt]);
        }
    }
    int col = lane & 15;
#pragma unroll
    for (int v = 0; v < 8; ++v) {
        int mr = mTile * 16 + v + ((lane >> 4) << 3);
        int nImg = mr / HW;
        int pos  = mr - nImg * HW;
        if (col < AAN) {
            float logit = acc[0][v] + cls_b[col];
            scores[(size_t)nImg * ANCH + pos * AAN + col] = 1.f / (1.f + __expf(-logit));
        }
#pragma unroll
        for (int bt = 0; bt < 4; ++bt) {
            int c = bt * 16 + col;
            if (c < 60) {
                float val = acc[1 + bt][v] + bbox_b[c];
                int aIdx = c >> 2, d = c & 3;
                breg[(((size_t)nImg * ANCH + pos * AAN + aIdx) << 2) + d] = val;
            }
        }
    }
}

// ---------------- 2-level radix select (top-2000 per image) ----------------
__global__ void k_hist1(const float* __restrict__ scores, unsigned* __restrict__ hist) {
    int g = blockIdx.x * blockDim.x + threadIdx.x;
    if (g >= NB * ANCH) return;
    int n = g / ANCH;
    unsigned key = __float_as_uint(scores[g]);   // positive floats: bit-order == value-order
    atomicAdd(&hist[n * 2048 + (key >> 21)], 1u);
}

__global__ void k_scan1(const unsigned* __restrict__ hist, int* __restrict__ meta) {
    if (threadIdx.x != 0) return;
    int n = blockIdx.x;
    int cum = 0, b1 = 0, above = 0;
    for (int b = 2047; b >= 0; --b) {
        int h = (int)hist[n * 2048 + b];
        if (cum + h >= PRE) { b1 = b; above = cum; break; }
        cum += h;
    }
    meta[n * 8 + 0] = b1;
    meta[n * 8 + 1] = above;
}

__global__ void k_hist2(const float* __restrict__ scores, const int* __restrict__ meta,
                        unsigned* __restrict__ hist2) {
    int g = blockIdx.x * blockDim.x + threadIdx.x;
    if (g >= NB * ANCH) return;
    int n = g / ANCH;
    unsigned key = __float_as_uint(scores[g]);
    if ((int)(key >> 21) == meta[n * 8 + 0])
        atomicAdd(&hist2[n * 2048 + ((key >> 10) & 0x7FF)], 1u);
}

__global__ void k_scan2(const unsigned* __restrict__ hist2, int* __restrict__ meta) {
    if (threadIdx.x != 0) return;
    int n = blockIdx.x;
    int cum = meta[n * 8 + 1];
    int b2 = 0;
    for (int b = 2047; b >= 0; --b) {
        cum += (int)hist2[n * 2048 + b];
        if (cum >= PRE) { b2 = b; break; }
    }
    meta[n * 8 + 2] = (meta[n * 8 + 0] << 21) | (b2 << 10);   // threshold key
}

__global__ void k_collect(const float* __restrict__ scores, int* __restrict__ meta,
                          unsigned long long* __restrict__ pairs) {
    int g = blockIdx.x * blockDim.x + threadIdx.x;
    if (g >= NB * ANCH) return;
    int n = g / ANCH;
    int i = g - n * ANCH;
    unsigned key = __float_as_uint(scores[g]);
    if (key >= (unsigned)meta[n * 8 + 2]) {
        int p = atomicAdd(&meta[n * 8 + 3], 1);
        if (p < CAND_MAX)
            pairs[(size_t)n * CAND_MAX + p] = ((unsigned long long)key << 32) | (unsigned)i;
    }
}

// ---------------- sort + decode + clip + NMS + compact (one WG per image) ----------------
__global__ __launch_bounds__(1024) void k_sortnms(const unsigned long long* __restrict__ gpairs,
                                                  const int* __restrict__ meta,
                                                  const float* __restrict__ breg,
                                                  const float* __restrict__ anchors,
                                                  const int* __restrict__ img_shapes,
                                                  float* __restrict__ out) {
    __shared__ union SU {
        unsigned long long pairs[CAND_MAX];  // phase 1: bitonic sort
        float boxes[PRE * 4];                // phase 2: decoded boxes
    } u;
    __shared__ int selidx[PRE];
    __shared__ int state[PRE];               // 0 = valid & kept so far

    int n = blockIdx.x;
    int tid = threadIdx.x;
    int cnt = meta[n * 8 + 3];
    if (cnt > CAND_MAX) cnt = CAND_MAX;

    for (int i = tid; i < CAND_MAX; i += 1024)
        u.pairs[i] = (i < cnt) ? gpairs[(size_t)n * CAND_MAX + i] : 0ull;
    __syncthreads();

    // bitonic sort, descending
    for (int k = 2; k <= CAND_MAX; k <<= 1) {
        for (int j = k >> 1; j > 0; j >>= 1) {
            for (int i = tid; i < CAND_MAX; i += 1024) {
                int ixj = i ^ j;
                if (ixj > i) {
                    unsigned long long a = u.pairs[i], b = u.pairs[ixj];
                    bool desc = ((i & k) == 0);
                    if (desc ? (a < b) : (a > b)) { u.pairs[i] = b; u.pairs[ixj] = a; }
                }
            }
            __syncthreads();
        }
    }

    for (int i = tid; i < PRE; i += 1024)
        selidx[i] = (i < cnt) ? (int)(u.pairs[i] & 0xFFFFFFFFull) : -1;
    __syncthreads();   // pairs region dead -> reuse as boxes

    float w_img = (float)img_shapes[n * 2 + 1];
    float h_img = (float)img_shapes[n * 2 + 0];
    for (int i = tid; i < PRE; i += 1024) {
        int idx = selidx[i];
        float x1 = 0.f, y1 = 0.f, x2 = 0.f, y2 = 0.f;
        int st = 1;
        if (idx >= 0) {
            const float* r = breg + (((size_t)n * ANCH + idx) << 2);
            const float* a = anchors + ((size_t)idx << 2);
            float wa = a[2] - a[0], ha = a[3] - a[1];
            float cxa = a[0] + 0.5f * wa, cya = a[1] + 0.5f * ha;
            float dw = fminf(r[2], XFORM_CLIP), dh = fminf(r[3], XFORM_CLIP);
            float cx = r[0] * wa + cxa, cy = r[1] * ha + cya;
            float pw = __expf(dw) * wa, ph = __expf(dh) * ha;
            x1 = cx - 0.5f * pw; y1 = cy - 0.5f * ph;
            x2 = cx + 0.5f * pw; y2 = cy + 0.5f * ph;
            x1 = fminf(fmaxf(x1, 0.f), w_img); x2 = fminf(fmaxf(x2, 0.f), w_img);
            y1 = fminf(fmaxf(y1, 0.f), h_img); y2 = fminf(fmaxf(y2, 0.f), h_img);
            if ((x2 - x1 >= 1.f) && (y2 - y1 >= 1.f)) st = 0;
        }
        u.boxes[i * 4 + 0] = x1; u.boxes[i * 4 + 1] = y1;
        u.boxes[i * 4 + 2] = x2; u.boxes[i * 4 + 3] = y2;
        state[i] = st;
    }
    __syncthreads();

    // greedy NMS (sequential over i, parallel over j)
    for (int i = 0; i < PRE; ++i) {
        if (state[i] == 0) {
            float bx1 = u.boxes[i * 4 + 0], by1 = u.boxes[i * 4 + 1];
            float bx2 = u.boxes[i * 4 + 2], by2 = u.boxes[i * 4 + 3];
            float area_i = fmaxf(bx2 - bx1, 0.f) * fmaxf(by2 - by1, 0.f);
            for (int j = i + 1 + tid; j < PRE; j += 1024) {
                float cx1 = u.boxes[j * 4 + 0], cy1 = u.boxes[j * 4 + 1];
                float cx2 = u.boxes[j * 4 + 2], cy2 = u.boxes[j * 4 + 3];
                float area_j = fmaxf(cx2 - cx1, 0.f) * fmaxf(cy2 - cy1, 0.f);
                float iw = fmaxf(fminf(bx2, cx2) - fmaxf(bx1, cx1), 0.f);
                float ih = fmaxf(fminf(by2, cy2) - fmaxf(by1, cy1), 0.f);
                float inter = iw * ih;
                float iou = inter / (area_i + area_j - inter + 1e-9f);
                if (iou > NMS_TH) state[j] |= 2;
            }
        }
        __syncthreads();
    }

    if (tid == 0) {
        int rank = 0;
        for (int i = 0; i < PRE && rank < POST; ++i) {
            if (state[i] == 0) {
                float* o = out + (((size_t)n * POST + rank) << 2);
                o[0] = u.boxes[i * 4 + 0]; o[1] = u.boxes[i * 4 + 1];
                o[2] = u.boxes[i * 4 + 2]; o[3] = u.boxes[i * 4 + 3];
                ++rank;
            }
        }
    }
}

// ---------------- host launcher ----------------
extern "C" void kernel_launch(void* const* d_in, const int* in_sizes, int n_in,
                              void* d_out, int out_size, void* d_ws, size_t ws_size,
                              hipStream_t stream) {
    (void)in_sizes; (void)n_in; (void)out_size; (void)ws_size;
    const float* features   = (const float*)d_in[0];
    const float* conv_w     = (const float*)d_in[1];
    const float* conv_b     = (const float*)d_in[2];
    const float* cls_w      = (const float*)d_in[3];
    const float* cls_b      = (const float*)d_in[4];
    const float* bbox_w     = (const float*)d_in[5];
    const float* bbox_b     = (const float*)d_in[6];
    const float* anchors    = (const float*)d_in[7];
    const int*   img_shapes = (const int*)d_in[8];
    float* out = (float*)d_out;

    char* ws = (char*)d_ws;
    size_t off = 0;
    auto take = [&](size_t bytes) -> char* {
        char* p = ws + off;
        off += (bytes + 255) & ~(size_t)255;
        return p;
    };
    unsigned short* featB  = (unsigned short*)take((size_t)NB * HW * 256 * 2);  // 31.1 MB
    unsigned short* tB     = (unsigned short*)take((size_t)NB * HW * 256 * 2);  // 31.1 MB
    unsigned short* wpack  = (unsigned short*)take((size_t)KSTEPS_CONV * COTILES * 512 * 2);
    unsigned short* wheads = (unsigned short*)take((size_t)WHEADS_BYTES);       // cls+bbox packed
    float* scores = (float*)take((size_t)NB * ANCH * 4);                        // 3.65 MB
    float* breg   = (float*)take((size_t)NB * ANCH * 4 * 4);                    // 14.6 MB
    unsigned* hist1 = (unsigned*)take((size_t)NB * 2048 * 4);
    unsigned* hist2 = (unsigned*)take((size_t)NB * 2048 * 4);
    int* meta = (int*)take((size_t)NB * 8 * 4);
    unsigned long long* pairs = (unsigned long long*)take((size_t)NB * CAND_MAX * 8);

    // zero histograms + meta (contiguous) and the output
    int histWords = (NB * 2048 * 2) + ((NB * 8 * 4 + 255) / 4);
    k_zero32<<<(histWords + 255) / 256, 256, 0, stream>>>(hist1, histWords);
    k_zero32<<<(NB * POST * 4 + 255) / 256, 256, 0, stream>>>((unsigned*)out, NB * POST * 4);

    int elemsF = NB * HW * 256;
    k_cvt_feat<<<(elemsF + 255) / 256, 256, 0, stream>>>(features, featB);
    k_pack_conv<<<(KSTEPS_CONV * COTILES * 512 + 255) / 256, 256, 0, stream>>>(conv_w, wpack);
    k_pack_heads<<<((8 + 32) * 512 + 255) / 256, 256, 0, stream>>>(cls_w, bbox_w, wheads);

    // conv: 3800 * 4 waves (16x64 tile per wave), 8 waves per block
    k_conv3x3<<<(MTILES * 4) / 8, 256, 0, stream>>>(featB, wpack, conv_b, tB);
    // heads: 3800 waves, 8 per block; TDM stages weights into LDS
    k_heads<<<MTILES / 8, 256, 0, stream>>>(tB, wheads, cls_b, bbox_b, scores, breg);

    int elemsS = NB * ANCH;
    int blkS = (elemsS + 255) / 256;
    k_hist1<<<blkS, 256, 0, stream>>>(scores, hist1);
    k_scan1<<<NB, 32, 0, stream>>>(hist1, meta);
    k_hist2<<<blkS, 256, 0, stream>>>(scores, meta, hist2);
    k_scan2<<<NB, 32, 0, stream>>>(hist2, meta);
    k_collect<<<blkS, 256, 0, stream>>>(scores, meta, pairs);

    k_sortnms<<<NB, 1024, 0, stream>>>(pairs, meta, breg, anchors, img_shapes, out);
}
